// PartAttentionBlock_54211077210190
// MI455X (gfx1250) — compile-verified, matrix operand order checked by reading, not run on hardware
//
#include <hip/hip_runtime.h>
#include <hip/hip_bf16.h>

typedef __bf16 bf16_t;
typedef __attribute__((ext_vector_type(16))) __bf16 v16bf;
typedef __attribute__((ext_vector_type(8)))  __bf16 v8bf;
typedef __attribute__((ext_vector_type(8)))  float  v8f;

// ---- problem constants ----
#define Bd 8
#define Fd 512
#define Kd 64
#define EDd 128
#define Wd 16
#define TPd 4
#define SHIFTd 2
#define Hd 4
#define Ld 64
#define fDd 128           // F / TP
#define nWd 4             // K / W
#define SCALEf 0.17677669529663687f   // (ED/H)^-0.5 = 1/sqrt(32)

// ---- WMMA helpers ----------------------------------------------------------
__device__ __forceinline__ v8f wmma_bf16(v16bf a, v16bf b, v8f c) {
  // D = A(16x32 bf16) * B(32x16 bf16) + C(16x16 f32)
  return __builtin_amdgcn_wmma_f32_16x16x32_bf16(
      /*neg_a=*/false, a, /*neg_b=*/false, b,
      /*c_mod=*/(short)0, c, /*reuse_a=*/false, /*reuse_b=*/false);
}

// Load a 16x32 bf16 fragment (A layout; B mirrored with N as "row").
// CDNA5 A layout: lane holds row M = lane&15; lane group g = lane>>4.
// elements 0..7  -> K = k0 + 8g + 0..7   (one contiguous 16B load)
// elements 8..15 -> K = k0 + 8g + 16..23 (one contiguous 16B load)
__device__ __forceinline__ v16bf load_frag16(const bf16_t* base, int row0,
                                             int ld, int k0) {
  const int lane = threadIdx.x & 31;
  const int r = lane & 15, g = lane >> 4;
  const bf16_t* p = base + (size_t)(row0 + r) * ld + k0 + g * 8;
  v8bf lo = *(const v8bf*)p;
  v8bf hi = *(const v8bf*)(p + 16);
  v16bf out;
#pragma unroll
  for (int i = 0; i < 8; ++i) { out[i] = lo[i]; out[i + 8] = hi[i]; }
  return out;
}

// ---- kernel 0: weight fp32 -> bf16 (weights then live in L2) --------------
__global__ void __launch_bounds__(256) cvt_weights_kernel(
    const float* __restrict__ qkv_w, const float* __restrict__ proj_w,
    const float* __restrict__ fc1_w, const float* __restrict__ fc2_w,
    bf16_t* __restrict__ ws) {
  int i = blockIdx.x * 256 + threadIdx.x;
  if (i < 49152)        ws[i] = (bf16_t)qkv_w[i];
  else if (i < 65536)   ws[i] = (bf16_t)proj_w[i - 49152];
  else if (i < 131072)  ws[i] = (bf16_t)fc1_w[i - 65536];
  else if (i < 196608)  ws[i] = (bf16_t)fc2_w[i - 131072];
}

// ---- kernel 1: shifted-window attention (one block per window) ------------
// LDS layout (phase-aliased), strides padded to multiples of 8 halves:
//   [0      ,17408)  qS  [64][136] bf16   (QKV..S)      | pS [4][64][72] (softmax..PV)
//   [17408  ,34816)  kS  [64][136] bf16   (QKV..S)      | (pS cont.)
//   [0      ,33792)  prS [64][132] f32    (proj..out)
//   [36864  ,54272)  oS  [64][136] bf16   (PV..proj)
//   [54272  ,72704)  vS  [128][72] bf16 (vT)  (QKV..PV)
//   [72704  ,90112)  xnS [64][136] bf16   (LN..QKV)
//   [72704  ,140288) sS  [4][64][66] f32  (S..softmax)
__global__ void __launch_bounds__(256) attn_kernel(
    const float* __restrict__ x, const bf16_t* __restrict__ qkvW,
    const float* __restrict__ qkv_b, const bf16_t* __restrict__ projW,
    const float* __restrict__ proj_b, const float* __restrict__ ln1_s,
    const float* __restrict__ ln1_b, float* __restrict__ out) {
  extern __shared__ char smem[];
  bf16_t* qS  = (bf16_t*)(smem + 0);
  bf16_t* kS  = (bf16_t*)(smem + 17408);
  bf16_t* pS  = (bf16_t*)(smem + 0);
  float*  prS = (float*)(smem + 0);
  bf16_t* oS  = (bf16_t*)(smem + 36864);
  bf16_t* vS  = (bf16_t*)(smem + 54272);
  bf16_t* xnS = (bf16_t*)(smem + 72704);
  float*  sS  = (float*)(smem + 72704);

  const int tid = threadIdx.x;
  const int lane = tid & 31;
  const int wv = tid >> 5;
  const int n = blockIdx.x;            // window id = ((b*128 + fi)*4 + wi)
  const int wi = n & 3;
  const int fi = (n >> 2) & 127;
  const int b  = n >> 9;

  // ---- phase A: gather window (roll -SHIFT along F) + LayerNorm1 ----------
  {
    const int tok = tid >> 2;          // 0..63, 4 threads per token
    const int part = tid & 3;          // 32 channels each
    const int tp = tok >> 4, w = tok & 15;
    const int fsrc = (fi * TPd + tp + SHIFTd) & (Fd - 1);
    const int kidx = wi * Wd + w;
    const float* src =
        x + (((size_t)b * Fd + fsrc) * Kd + kidx) * EDd + part * 32;
    float4 v[8];
    float s = 0.f, s2 = 0.f;
#pragma unroll
    for (int i = 0; i < 8; ++i) {
      v[i] = ((const float4*)src)[i];
      s  += v[i].x + v[i].y + v[i].z + v[i].w;
      s2 += v[i].x * v[i].x + v[i].y * v[i].y + v[i].z * v[i].z + v[i].w * v[i].w;
    }
    s  += __shfl_xor(s, 1);  s  += __shfl_xor(s, 2);
    s2 += __shfl_xor(s2, 1); s2 += __shfl_xor(s2, 2);
    const float mean = s * (1.f / 128.f);
    const float var  = s2 * (1.f / 128.f) - mean * mean;
    const float rstd = rsqrtf(var + 1e-5f);
#pragma unroll
    for (int i = 0; i < 8; ++i) {
      const int c = part * 32 + i * 4;
      const float4 g  = *(const float4*)(ln1_s + c);
      const float4 be = *(const float4*)(ln1_b + c);
      bf16_t* dst = xnS + tok * 136 + c;
      dst[0] = (bf16_t)((v[i].x - mean) * rstd * g.x + be.x);
      dst[1] = (bf16_t)((v[i].y - mean) * rstd * g.y + be.y);
      dst[2] = (bf16_t)((v[i].z - mean) * rstd * g.z + be.z);
      dst[3] = (bf16_t)((v[i].w - mean) * rstd * g.w + be.w);
    }
  }
  __syncthreads();

  // ---- phase B: QKV GEMM (64x128)@(128x384)^T ----------------------------
  // Tiles split into three uniform, branch-free store loops:
  //   t 0..31  -> q (scaled), t 32..63 -> k, t 64..95 -> v (transposed).
  for (int t = wv; t < 32; t += 8) {           // ---- q ----
    const int m = t & 3, nn = t >> 2;          // nn 0..7
    if (t + 8 < 32) __builtin_prefetch(qkvW + (size_t)((t + 8) >> 2) * 16 * 128, 0, 3);
    v8f acc = {};
#pragma unroll
    for (int kk = 0; kk < 4; ++kk) {
      v16bf a  = load_frag16(xnS, m * 16, 136, kk * 32);
      v16bf bb = load_frag16(qkvW, nn * 16, 128, kk * 32);
      acc = wmma_bf16(a, bb, acc);
    }
    const int col = nn * 16 + (lane & 15);
    const int rbase = m * 16 + (lane >> 4) * 8;
    const float bias = qkv_b[col];
#pragma unroll
    for (int r = 0; r < 8; ++r)
      qS[(rbase + r) * 136 + col] = (bf16_t)((acc[r] + bias) * SCALEf);
  }
  for (int t = wv; t < 32; t += 8) {           // ---- k ----
    const int m = t & 3, nn = 8 + (t >> 2);    // nn 8..15
    if (t + 8 < 32) __builtin_prefetch(qkvW + (size_t)(8 + ((t + 8) >> 2)) * 16 * 128, 0, 3);
    v8f acc = {};
#pragma unroll
    for (int kk = 0; kk < 4; ++kk) {
      v16bf a  = load_frag16(xnS, m * 16, 136, kk * 32);
      v16bf bb = load_frag16(qkvW, nn * 16, 128, kk * 32);
      acc = wmma_bf16(a, bb, acc);
    }
    const int col = nn * 16 + (lane & 15);
    const int rbase = m * 16 + (lane >> 4) * 8;
    const float bias = qkv_b[col];
#pragma unroll
    for (int r = 0; r < 8; ++r)
      kS[(rbase + r) * 136 + (col - 128)] = (bf16_t)(acc[r] + bias);
  }
  for (int t = wv; t < 32; t += 8) {           // ---- v (store transposed) --
    const int m = t & 3, nn = 16 + (t >> 2);   // nn 16..23
    if (t + 8 < 32) __builtin_prefetch(qkvW + (size_t)(16 + ((t + 8) >> 2)) * 16 * 128, 0, 3);
    v8f acc = {};
#pragma unroll
    for (int kk = 0; kk < 4; ++kk) {
      v16bf a  = load_frag16(xnS, m * 16, 136, kk * 32);
      v16bf bb = load_frag16(qkvW, nn * 16, 128, kk * 32);
      acc = wmma_bf16(a, bb, acc);
    }
    const int col = nn * 16 + (lane & 15);
    const int rbase = m * 16 + (lane >> 4) * 8;
    const float bias = qkv_b[col];
#pragma unroll
    for (int r = 0; r < 8; ++r)
      vS[(col - 256) * 72 + (rbase + r)] = (bf16_t)(acc[r] + bias);
  }
  __syncthreads();

  // ---- phase C: S = q k^T per head (Dh=32 => single WMMA per tile) --------
  for (int t = wv; t < 64; t += 8) {
    const int h = t >> 4, sub = t & 15;
    const int ti = sub >> 2, tj = sub & 3;
    v16bf a  = load_frag16(qS, ti * 16, 136, h * 32);
    v16bf bb = load_frag16(kS, tj * 16, 136, h * 32);
    v8f acc = {};
    acc = wmma_bf16(a, bb, acc);
    const int col = tj * 16 + (lane & 15);
    const int rbase = ti * 16 + (lane >> 4) * 8;
    float* Sh = sS + h * 64 * 66;
#pragma unroll
    for (int r = 0; r < 8; ++r) Sh[(rbase + r) * 66 + col] = acc[r];
  }
  __syncthreads();

  // ---- phase D: mask + masked_fill + softmax (one row per thread) ---------
  {
    const int h = tid >> 6, i = tid & 63;
    const bool lastf = (fi == fDd - 1);
    const int mi = lastf ? (((i >> 4) < 2) ? 1 : 2) : 0;
    const float* Sr = sS + (h * 64 + i) * 66;
    float vals[64];
    float mx = -3.4e38f;
#pragma unroll
    for (int j = 0; j < 64; ++j) {
      const int mj = lastf ? (((j >> 4) < 2) ? 1 : 2) : 0;
      const float sv = Sr[j];
      const float a = (mi == mj && sv != 0.f) ? sv : -10000.f;
      vals[j] = a;
      mx = fmaxf(mx, a);
    }
    float sum = 0.f;
#pragma unroll
    for (int j = 0; j < 64; ++j) { vals[j] = __expf(vals[j] - mx); sum += vals[j]; }
    const float inv = 1.f / sum;
    bf16_t* Pr = pS + (h * 64 + i) * 72;
#pragma unroll
    for (int j = 0; j < 64; ++j) Pr[j] = (bf16_t)(vals[j] * inv);
  }
  __syncthreads();

  // ---- phase E: O = P @ V per head ----------------------------------------
  for (int t = wv; t < 32; t += 8) {
    const int h = t >> 3, sub = t & 7;
    const int ti = sub >> 1, tj = sub & 1;
    v8f acc = {};
#pragma unroll
    for (int kk = 0; kk < 2; ++kk) {
      v16bf a  = load_frag16(pS + h * 64 * 72, ti * 16, 72, kk * 32);
      v16bf bb = load_frag16(vS, h * 32 + tj * 16, 72, kk * 32);  // vT rows
      acc = wmma_bf16(a, bb, acc);
    }
    const int col = h * 32 + tj * 16 + (lane & 15);
    const int rbase = ti * 16 + (lane >> 4) * 8;
#pragma unroll
    for (int r = 0; r < 8; ++r) oS[(rbase + r) * 136 + col] = (bf16_t)acc[r];
  }
  __syncthreads();

  // ---- phase F: proj GEMM (64x128)@(128x128)^T ----------------------------
  for (int t = wv; t < 32; t += 8) {
    const int m = t & 3, nn = t >> 2;
    if (t + 8 < 32) __builtin_prefetch(projW + (size_t)((t + 8) >> 2) * 16 * 128, 0, 3);
    v8f acc = {};
#pragma unroll
    for (int kk = 0; kk < 4; ++kk) {
      v16bf a  = load_frag16(oS, m * 16, 136, kk * 32);
      v16bf bb = load_frag16(projW, nn * 16, 128, kk * 32);
      acc = wmma_bf16(a, bb, acc);
    }
    const int col = nn * 16 + (lane & 15);
    const int rbase = m * 16 + (lane >> 4) * 8;
    const float bias = proj_b[col];
#pragma unroll
    for (int r = 0; r < 8; ++r) prS[(rbase + r) * 132 + col] = acc[r] + bias;
  }
  __syncthreads();

  // ---- phase G: roll(+SHIFT) over tokens, window_reverse, + residual ------
  for (int idx = tid; idx < 64 * 32; idx += 256) {
    const int ld = idx >> 5;                 // destination token
    const int c4 = (idx & 31) * 4;
    const int ls = (ld - SHIFTd + 64) & 63;  // rolled source token
    const int tp = ld >> 4, w = ld & 15;
    const int fidx = fi * TPd + tp, kidx = wi * Wd + w;
    const size_t g = (((size_t)b * Fd + fidx) * Kd + kidx) * EDd + c4;
    float4 pv = *(float4*)(prS + ls * 132 + c4);
    const float4 sc = *(const float4*)(x + g);
    pv.x += sc.x; pv.y += sc.y; pv.z += sc.z; pv.w += sc.w;
    *(float4*)(out + g) = pv;
  }
}

// ---- kernel 2: LN2 + MLP (fc1/GELU/fc2) + residual, 64 tokens/block ------
// LDS: xf [64][132] f32 @0 (33792) | xn [64][136] bf16 @33792 (17408)
//      h1 [64][520] bf16 @51200 (66560)   total 117760 B
__global__ void __launch_bounds__(256) mlp_kernel(
    const float* __restrict__ ln2_s, const float* __restrict__ ln2_b,
    const bf16_t* __restrict__ fc1W, const float* __restrict__ fc1_b,
    const bf16_t* __restrict__ fc2W, const float* __restrict__ fc2_b,
    float* __restrict__ io) {
  extern __shared__ char smem[];
  float*  xf = (float*)smem;
  bf16_t* xn = (bf16_t*)(smem + 33792);
  bf16_t* h1 = (bf16_t*)(smem + 51200);

  const int tid = threadIdx.x;
  const int lane = tid & 31;
  const int wv = tid >> 5;
  const size_t t0 = (size_t)blockIdx.x * 64;

  // ---- LN2 (keep raw x in LDS for the residual) ---------------------------
  {
    const int tok = tid >> 2, part = tid & 3;
    const float* src = io + (t0 + tok) * EDd + part * 32;
    float4 v[8];
    float s = 0.f, s2 = 0.f;
#pragma unroll
    for (int i = 0; i < 8; ++i) {
      v[i] = ((const float4*)src)[i];
      s  += v[i].x + v[i].y + v[i].z + v[i].w;
      s2 += v[i].x * v[i].x + v[i].y * v[i].y + v[i].z * v[i].z + v[i].w * v[i].w;
      *(float4*)(xf + tok * 132 + part * 32 + i * 4) = v[i];
    }
    s  += __shfl_xor(s, 1);  s  += __shfl_xor(s, 2);
    s2 += __shfl_xor(s2, 1); s2 += __shfl_xor(s2, 2);
    const float mean = s * (1.f / 128.f);
    const float var  = s2 * (1.f / 128.f) - mean * mean;
    const float rstd = rsqrtf(var + 1e-5f);
#pragma unroll
    for (int i = 0; i < 8; ++i) {
      const int c = part * 32 + i * 4;
      const float4 g  = *(const float4*)(ln2_s + c);
      const float4 be = *(const float4*)(ln2_b + c);
      bf16_t* dst = xn + tok * 136 + c;
      dst[0] = (bf16_t)((v[i].x - mean) * rstd * g.x + be.x);
      dst[1] = (bf16_t)((v[i].y - mean) * rstd * g.y + be.y);
      dst[2] = (bf16_t)((v[i].z - mean) * rstd * g.z + be.z);
      dst[3] = (bf16_t)((v[i].w - mean) * rstd * g.w + be.w);
    }
  }
  __syncthreads();

  // ---- fc1 + exact GELU: (64x128)@(128x512)^T -----------------------------
  for (int t = wv; t < 128; t += 8) {
    const int m = t & 3, nn = t >> 2;     // 4 x 32 tiles
    if (t + 8 < 128) __builtin_prefetch(fc1W + (size_t)((t + 8) >> 2) * 16 * 128, 0, 3);
    v8f acc = {};
#pragma unroll
    for (int kk = 0; kk < 4; ++kk) {
      v16bf a  = load_frag16(xn, m * 16, 136, kk * 32);
      v16bf bb = load_frag16(fc1W, nn * 16, 128, kk * 32);
      acc = wmma_bf16(a, bb, acc);
    }
    const int col = nn * 16 + (lane & 15);
    const int rbase = m * 16 + (lane >> 4) * 8;
    const float bias = fc1_b[col];
#pragma unroll
    for (int r = 0; r < 8; ++r) {
      const float u = acc[r] + bias;
      const float ge = 0.5f * u * (1.f + erff(u * 0.70710678118654752f));
      h1[(rbase + r) * 520 + col] = (bf16_t)ge;
    }
  }
  __syncthreads();

  // ---- fc2 + residual: (64x512)@(512x128)^T -------------------------------
  for (int t = wv; t < 32; t += 8) {
    const int m = t & 3, nn = t >> 2;     // 4 x 8 tiles
    if (t + 8 < 32) __builtin_prefetch(fc2W + (size_t)((t + 8) >> 2) * 16 * 512, 0, 3);
    v8f acc = {};
#pragma unroll
    for (int kk = 0; kk < 16; ++kk) {
      v16bf a  = load_frag16(h1, m * 16, 520, kk * 32);
      v16bf bb = load_frag16(fc2W, nn * 16, 512, kk * 32);
      acc = wmma_bf16(a, bb, acc);
    }
    const int col = nn * 16 + (lane & 15);
    const int rbase = m * 16 + (lane >> 4) * 8;
    const float bias = fc2_b[col];
#pragma unroll
    for (int r = 0; r < 8; ++r) {
      const int row = rbase + r;
      io[(t0 + row) * EDd + col] = acc[r] + bias + xf[row * 132 + col];
    }
  }
}

// ---- host launcher --------------------------------------------------------
extern "C" void kernel_launch(void* const* d_in, const int* in_sizes, int n_in,
                              void* d_out, int out_size, void* d_ws,
                              size_t ws_size, hipStream_t stream) {
  (void)in_sizes; (void)n_in; (void)out_size; (void)ws_size;
  const float* x      = (const float*)d_in[0];
  const float* qkv_w  = (const float*)d_in[1];
  const float* qkv_b  = (const float*)d_in[2];
  const float* proj_w = (const float*)d_in[3];
  const float* proj_b = (const float*)d_in[4];
  const float* ln1_s  = (const float*)d_in[5];
  const float* ln1_b  = (const float*)d_in[6];
  const float* ln2_s  = (const float*)d_in[7];
  const float* ln2_b  = (const float*)d_in[8];
  const float* fc1_w  = (const float*)d_in[9];
  const float* fc1_b  = (const float*)d_in[10];
  const float* fc2_w  = (const float*)d_in[11];
  const float* fc2_b  = (const float*)d_in[12];
  float* out = (float*)d_out;

  bf16_t* wsb   = (bf16_t*)d_ws;
  bf16_t* qkvW  = wsb;                 // [384][128]
  bf16_t* projW = wsb + 49152;         // [128][128]
  bf16_t* fc1W  = wsb + 65536;         // [512][128]
  bf16_t* fc2W  = wsb + 131072;        // [128][512]

  cvt_weights_kernel<<<768, 256, 0, stream>>>(qkv_w, proj_w, fc1_w, fc2_w, wsb);

  // one block per attention window: B * (F/TP) * (K/W) = 8*128*4 = 4096
  attn_kernel<<<4096, 256, 140288, stream>>>(x, qkvW, qkv_b, projW, proj_b,
                                             ln1_s, ln1_b, out);

  // 262144 tokens / 64 per block = 4096 blocks
  mlp_kernel<<<4096, 256, 117760, stream>>>(ln2_s, ln2_b, fc1W, fc1_b, fc2W,
                                            fc2_b, out);
}
